// PointNet2Msg_62414464746067
// MI455X (gfx1250) — matrix operand order repaired and checked by reading, not compiled, not run on hardware
//
#include <hip/hip_runtime.h>

// ---------------------------------------------------------------------------
// PointNet++ MSG forward for MI455X (gfx1250).
// MLP GEMMs: v_wmma_f32_16x16x32_bf16, wave32, 64x32 block tile (4 waves,
// each wave 16x32 via two WMMAs sharing the A fragment). All GEMM K dims are
// zero-padded to multiples of 32 -> guard-free aligned b128 staging.
// BatchNorm is fused: the GEMM epilogue emits deterministic per-block channel
// sums/sumsq (fixed-order LDS reduction), bn_finalize turns them into
// scale/shift, and BN+ReLU is applied by consumers (next GEMM's A staging,
// maxpool, copies, head) -- one write + one read per activation tensor.
// Activations/weights stored bf16 to halve HBM/L2 traffic.
// ---------------------------------------------------------------------------

#define BB 8
#define NN 4096
#define S1P 512
#define S2P 128

typedef __bf16 bf16x16 __attribute__((ext_vector_type(16)));
typedef float f32x8 __attribute__((ext_vector_type(8)));
typedef unsigned short u16x16 __attribute__((ext_vector_type(16)));
typedef unsigned short u16x8 __attribute__((ext_vector_type(8)));

static __device__ __forceinline__ unsigned short f2bf(float f) {
    union { float f; unsigned u; } v; v.f = f;
    unsigned b = v.u;
    unsigned r = (b + 0x7FFFu + ((b >> 16) & 1u)) >> 16;   // RNE
    return (unsigned short)r;
}
static __device__ __forceinline__ float bf2f(unsigned short h) {
    union { unsigned u; float f; } v; v.u = ((unsigned)h) << 16;
    return v.f;
}

// -------------------------- WMMA GEMM (fused BN) ---------------------------
// Y[M,Nc] = act(A)[M,Kp] * W[Kp,Nc] + bias, where act(x)=relu(scaleIn*x+
// shiftIn) per input channel (identity if scaleIn==nullptr). Also emits
// per-block per-channel partial sums/sumsq of Y (pre-bf16-round, f32) for the
// fused BatchNorm, reduced in a fixed order (deterministic).
// Preconditions: M%64==0, Kp%32==0, Nc%32==0, A/W 32B-aligned.
__global__ __launch_bounds__(128) void gemm_bias_wmma(
    const unsigned short* __restrict__ A,
    const unsigned short* __restrict__ Wm,
    const float* __restrict__ bias,
    const float* __restrict__ scaleIn,   // nullable: fused BN+ReLU on A
    const float* __restrict__ shiftIn,
    unsigned short* __restrict__ Y,
    float* __restrict__ partS,           // [gridX*Nc] per-block channel sums
    float* __restrict__ partQ,           // [gridX*Nc] per-block channel sumsq
    int M, int Kp, int Nc)
{
    __shared__ alignas(64) unsigned short lA[64][32];    // [m][k]
    __shared__ alignas(64) unsigned short lBt[32][32];   // [n][k] (transposed)
    __shared__ float psS[8][32];
    __shared__ float psQ[8][32];

    const int tid  = threadIdx.x;
    const int wave = tid >> 5;
    const int lane = tid & 31;
    const int rowBase = blockIdx.x * 64;
    const int nBase   = blockIdx.y * 32;

    // staging assignments (all aligned, no guards needed)
    const int arow  = tid >> 1;          // 0..63
    const int ahalf = (tid & 1) * 16;    // 0 or 16
    const int bk    = tid >> 2;          // 0..31  (k within tile)
    const int bq    = (tid & 3) * 8;     // 0,8,16,24 (n within tile)

    f32x8 acc0 = {}, acc1 = {};

    for (int k0 = 0; k0 < Kp; k0 += 32) {
        // A tile: one aligned 32B vector per thread, with optional fused
        // BN+ReLU of the producing layer applied in-flight.
        u16x16 ach = *(const u16x16*)(A + (size_t)(rowBase + arow) * Kp + (k0 + ahalf));
        if (scaleIn) {
            const float* sp = scaleIn + k0 + ahalf;
            const float* hp = shiftIn + k0 + ahalf;
            #pragma unroll
            for (int e = 0; e < 16; ++e)
                ach[e] = f2bf(fmaxf(sp[e] * bf2f(ach[e]) + hp[e], 0.f));
        }
        *(u16x16*)&lA[arow][ahalf] = ach;
        // B tile: aligned 16B vector per thread, scattered transposed to LDS
        u16x8 bch = *(const u16x8*)(Wm + (size_t)(k0 + bk) * Nc + (nBase + bq));
        #pragma unroll
        for (int e = 0; e < 8; ++e) lBt[bq + e][bk] = bch[e];
        __syncthreads();

        // A fragment: lane L -> row m=L&15; elems 0..15 = K kb..kb+15
        // with kb = (L>=16)?16:0.
        int m  = lane & 15;
        int kb = (lane >> 4) * 16;
        u16x8 a0 = *(const u16x8*)&lA[wave * 16 + m][kb];
        u16x8 a1 = *(const u16x8*)&lA[wave * 16 + m][kb + 8];
        u16x16 au;
        #pragma unroll
        for (int e = 0; e < 8; ++e) { au[e] = a0[e]; au[e + 8] = a1[e]; }

        // B fragments: lane L -> col n=L&15 (+16 for second tile);
        // elems e = K (L>=16?16:0)+e -> contiguous in transposed LDS.
        int n   = lane & 15;
        int kb2 = (lane >> 4) * 16;
        u16x16 b0u = *(const u16x16*)&lBt[n][kb2];
        u16x16 b1u = *(const u16x16*)&lBt[n + 16][kb2];

        bf16x16 av  = __builtin_bit_cast(bf16x16, au);
        bf16x16 bv0 = __builtin_bit_cast(bf16x16, b0u);
        bf16x16 bv1 = __builtin_bit_cast(bf16x16, b1u);
        acc0 = __builtin_amdgcn_wmma_f32_16x16x32_bf16(
                 false, av, false, bv0, (short)0, acc0, false, false);
        acc1 = __builtin_amdgcn_wmma_f32_16x16x32_bf16(
                 false, av, false, bv1, (short)0, acc1, false, false);
        __syncthreads();
    }

    // C/D layout: VGPR r: lanes0-15 -> M=r, lanes16-31 -> M=8+r; N = lane&15.
    int gn = nBase + (lane & 15);
    float bv0 = bias[gn], bv1 = bias[gn + 16];
    float s0 = 0.f, q0 = 0.f, s1 = 0.f, q1 = 0.f;
    #pragma unroll
    for (int r = 0; r < 8; ++r) {
        int gm = rowBase + wave * 16 + (lane >> 4) * 8 + r;
        float v0 = acc0[r] + bv0;
        float v1 = acc1[r] + bv1;
        Y[(size_t)gm * Nc + gn]      = f2bf(v0);
        Y[(size_t)gm * Nc + gn + 16] = f2bf(v1);
        s0 += v0; q0 += v0 * v0;
        s1 += v1; q1 += v1 * v1;
    }
    // fixed-order per-block channel reduction (deterministic, no atomics)
    int contrib = wave * 2 + (lane >> 4);
    int col = lane & 15;
    psS[contrib][col] = s0;       psQ[contrib][col] = q0;
    psS[contrib][col + 16] = s1;  psQ[contrib][col + 16] = q1;
    __syncthreads();
    if (tid < 32) {
        float s = 0.f, q = 0.f;
        #pragma unroll
        for (int j = 0; j < 8; ++j) { s += psS[j][tid]; q += psQ[j][tid]; }
        partS[(size_t)blockIdx.x * Nc + nBase + tid] = s;
        partQ[(size_t)blockIdx.x * Nc + nBase + tid] = q;
    }
}

// --------------------- BN finalize (partials -> scale/shift) ---------------
__global__ __launch_bounds__(256) void bn_finalize(
    const float* __restrict__ partS, const float* __restrict__ partQ,
    int nblocks, int Nc, int M,
    const float* __restrict__ gamma, const float* __restrict__ beta,
    float* __restrict__ scaleOut, float* __restrict__ shiftOut)
{
    int c = blockIdx.x;
    float s = 0.f, q = 0.f;
    for (int b = threadIdx.x; b < nblocks; b += 256) {
        s += partS[(size_t)b * Nc + c];
        q += partQ[(size_t)b * Nc + c];
    }
    __shared__ float sa[256], sb[256];
    sa[threadIdx.x] = s; sb[threadIdx.x] = q;
    __syncthreads();
    for (int st = 128; st > 0; st >>= 1) {
        if (threadIdx.x < st) {
            sa[threadIdx.x] += sa[threadIdx.x + st];
            sb[threadIdx.x] += sb[threadIdx.x + st];
        }
        __syncthreads();
    }
    if (threadIdx.x == 0) {
        float m = sa[0] / (float)M;
        float var = sb[0] / (float)M - m * m;
        float iv = rsqrtf(fmaxf(var, 0.f) + 1e-5f);
        float sc = gamma[c] * iv;
        scaleOut[c] = sc;
        shiftOut[c] = beta[c] - m * sc;
    }
}

// ----------------------------- helpers -------------------------------------
__global__ void transpose_in(const float* __restrict__ x, float* __restrict__ xyz,
                             int Npts, size_t total) {
    size_t i = (size_t)blockIdx.x * 256 + threadIdx.x;
    if (i >= total) return;
    size_t bn = i / 3; int d = (int)(i % 3);
    int b = (int)(bn / Npts), n = (int)(bn % Npts);
    xyz[i] = x[((size_t)b * 3 + d) * Npts + n];
}

// weight convert f32 -> bf16, zero-padding rows k >= K up to Kp
__global__ void conv_weight(const float* __restrict__ W, unsigned short* __restrict__ dst,
                            int K, int Nc, size_t total) {
    size_t i = (size_t)blockIdx.x * 256 + threadIdx.x;
    if (i >= total) return;
    int k = (int)(i / Nc), n = (int)(i % Nc);
    dst[i] = (k < K) ? f2bf(W[(size_t)k * Nc + n]) : (unsigned short)0;
}

__global__ __launch_bounds__(256) void fps_kernel(
    const float* __restrict__ xyz, int Npts, int npoint, int* __restrict__ outIdx)
{
    int b = blockIdx.x, tid = threadIdx.x;
    __shared__ float dist[NN];
    __shared__ float sv[256];
    __shared__ int   si[256];
    __shared__ int   curs;
    const float* base = xyz + (size_t)b * Npts * 3;
    for (int i = tid; i < Npts; i += 256) dist[i] = 1e10f;
    if (tid == 0) curs = 0;
    __syncthreads();
    for (int it = 0; it < npoint; ++it) {
        int cur = curs;
        if (tid == 0) outIdx[b * npoint + it] = cur;
        float cx = base[cur * 3], cy = base[cur * 3 + 1], cz = base[cur * 3 + 2];
        float best = -1.f; int bi = 0;
        for (int i = tid; i < Npts; i += 256) {
            float dx = base[i * 3] - cx, dy = base[i * 3 + 1] - cy, dz = base[i * 3 + 2] - cz;
            float d = dx * dx + dy * dy + dz * dz;
            float nd = fminf(dist[i], d);
            dist[i] = nd;
            if (nd > best) { best = nd; bi = i; }
        }
        sv[tid] = best; si[tid] = bi;
        __syncthreads();
        for (int st = 128; st > 0; st >>= 1) {
            if (tid < st) {
                if (sv[tid + st] > sv[tid] ||
                    (sv[tid + st] == sv[tid] && si[tid + st] < si[tid])) {
                    sv[tid] = sv[tid + st]; si[tid] = si[tid + st];
                }
            }
            __syncthreads();
        }
        if (tid == 0) curs = si[0];
        __syncthreads();
    }
}

__global__ void gather_xyz(const float* __restrict__ xyz, const int* __restrict__ fidx,
                           float* __restrict__ out, int Npts, int S, int total) {
    int g = blockIdx.x * 256 + threadIdx.x;
    if (g >= total) return;
    int b = g / S;
    int n = fidx[g];
    const float* p = &xyz[((size_t)b * Npts + n) * 3];
    out[(size_t)g * 3 + 0] = p[0];
    out[(size_t)g * 3 + 1] = p[1];
    out[(size_t)g * 3 + 2] = p[2];
}

__global__ void ball_query(const float* __restrict__ xyz, const float* __restrict__ nxyz,
                           int Npts, int S, float r2, int Kn, int* __restrict__ idx, int total) {
    int g = blockIdx.x * 256 + threadIdx.x;
    if (g >= total) return;
    int b = g / S;
    float cx = nxyz[(size_t)g * 3], cy = nxyz[(size_t)g * 3 + 1], cz = nxyz[(size_t)g * 3 + 2];
    int* out = &idx[(size_t)g * Kn];
    int cnt = 0;
    for (int n = 0; n < Npts && cnt < Kn; ++n) {
        const float* p = &xyz[((size_t)b * Npts + n) * 3];
        float dx = p[0] - cx, dy = p[1] - cy, dz = p[2] - cz;
        if (dx * dx + dy * dy + dz * dz <= r2) out[cnt++] = n;
    }
    int pad = (cnt > 0) ? out[0] : 0;
    for (; cnt < Kn; ++cnt) out[cnt] = pad;
}

// builds [rows, Cpad] rows of (feats, centered xyz, zero pad) in bf16
__global__ void group_kernel(const float* __restrict__ xyz, const float* __restrict__ nxyz,
                             const unsigned short* __restrict__ feats, const int* __restrict__ idx,
                             unsigned short* __restrict__ dst, int Npts, int S, int Kn,
                             int Cf, int Cpad, size_t total) {
    size_t i = (size_t)blockIdx.x * 256 + threadIdx.x;
    if (i >= total) return;
    int Cin = Cf + 3;
    size_t row = i / Cpad; int c = (int)(i % Cpad);
    if (c >= Cin) { dst[i] = 0; return; }
    size_t g = row / Kn;
    int b = (int)(g / S);
    int n = idx[row];
    if (c < Cf) {
        dst[i] = feats[((size_t)b * Npts + n) * Cf + c];
    } else {
        int d = c - Cf;
        dst[i] = f2bf(xyz[((size_t)b * Npts + n) * 3 + d] - nxyz[g * 3 + d]);
    }
}

// max over K of relu(scale*x+shift)  (applies the producing layer's BN)
__global__ void maxpool_k(const unsigned short* __restrict__ X,
                          const float* __restrict__ scale, const float* __restrict__ shift,
                          unsigned short* __restrict__ Y,
                          int Kn, int C, int outStride, int outOff, size_t total) {
    size_t i = (size_t)blockIdx.x * 256 + threadIdx.x;
    if (i >= total) return;
    size_t g = i / C; int c = (int)(i % C);
    float sc = scale[c], sh = shift[c];
    float m = -3e38f;
    for (int k = 0; k < Kn; ++k)
        m = fmaxf(m, fmaxf(sc * bf2f(X[(g * Kn + k) * C + c]) + sh, 0.f));
    Y[g * (size_t)outStride + outOff + c] = f2bf(m);
}

__global__ void concat_feats_xyz(const unsigned short* __restrict__ feats, int Cf,
                                 const float* __restrict__ xyz,
                                 unsigned short* __restrict__ dst, int Cpad, size_t total) {
    size_t i = (size_t)blockIdx.x * 256 + threadIdx.x;
    if (i >= total) return;
    int Cin = Cf + 3;
    size_t r = i / Cpad; int c = (int)(i % Cpad);
    if (c >= Cin) { dst[i] = 0; return; }
    dst[i] = (c < Cf) ? feats[r * Cf + c] : f2bf(xyz[r * 3 + (c - Cf)]);
}

__global__ void copy_bf16(const unsigned short* __restrict__ src, int srcStride,
                          unsigned short* __restrict__ dst, int dstStride, int dstOff,
                          int C, size_t total) {
    size_t i = (size_t)blockIdx.x * 256 + threadIdx.x;
    if (i >= total) return;
    size_t r = i / C; int c = (int)(i % C);
    dst[r * (size_t)dstStride + dstOff + c] = src[r * (size_t)srcStride + c];
}

// dst = relu(scale*src+shift), contiguous [rows, C]
__global__ void apply_copy(const unsigned short* __restrict__ src,
                           const float* __restrict__ scale, const float* __restrict__ shift,
                           unsigned short* __restrict__ dst, int C, size_t total) {
    size_t i = (size_t)blockIdx.x * 256 + threadIdx.x;
    if (i >= total) return;
    int c = (int)(i % C);
    dst[i] = f2bf(fmaxf(scale[c] * bf2f(src[i]) + shift[c], 0.f));
}

__global__ void bcast_rows(const unsigned short* __restrict__ src, unsigned short* __restrict__ dst,
                           int dstStride, int dstOff, int C, int per, size_t total) {
    size_t i = (size_t)blockIdx.x * 256 + threadIdx.x;
    if (i >= total) return;
    size_t r = i / C; int c = (int)(i % C);
    dst[r * (size_t)dstStride + dstOff + c] = src[(r / per) * (size_t)C + c];
}

__global__ void xyz_bf16(const float* __restrict__ xyz, unsigned short* __restrict__ dst,
                         int dstStride, int dstOff, size_t total) {
    size_t i = (size_t)blockIdx.x * 256 + threadIdx.x;
    if (i >= total) return;
    size_t r = i / 3; int c = (int)(i % 3);
    dst[r * (size_t)dstStride + dstOff + c] = f2bf(xyz[i]);
}

__global__ void zero_cols(unsigned short* __restrict__ dst, int stride, int off, int cnt,
                          size_t total) {
    size_t i = (size_t)blockIdx.x * 256 + threadIdx.x;
    if (i >= total) return;
    size_t r = i / cnt; int c = (int)(i % cnt);
    dst[r * (size_t)stride + off + c] = 0;
}

__global__ void three_nn_kernel(const float* __restrict__ q, const float* __restrict__ ref,
                                int N1, int S, int* __restrict__ idx3, float* __restrict__ w3,
                                int total) {
    int g = blockIdx.x * 256 + threadIdx.x;
    if (g >= total) return;
    int b = g / N1;
    float px = q[(size_t)g * 3], py = q[(size_t)g * 3 + 1], pz = q[(size_t)g * 3 + 2];
    float d0 = 1e30f, d1 = 1e30f, d2 = 1e30f;
    int i0 = 0, i1 = 0, i2 = 0;
    for (int s = 0; s < S; ++s) {
        const float* p = &ref[((size_t)b * S + s) * 3];
        float dx = p[0] - px, dy = p[1] - py, dz = p[2] - pz;
        float d = dx * dx + dy * dy + dz * dz;
        if (d < d0)      { d2 = d1; i2 = i1; d1 = d0; i1 = i0; d0 = d; i0 = s; }
        else if (d < d1) { d2 = d1; i2 = i1; d1 = d; i1 = s; }
        else if (d < d2) { d2 = d; i2 = s; }
    }
    float w0 = 1.f / (d0 + 1e-8f), w1 = 1.f / (d1 + 1e-8f), w2 = 1.f / (d2 + 1e-8f);
    float ws = w0 + w1 + w2;
    idx3[(size_t)g * 3 + 0] = i0; idx3[(size_t)g * 3 + 1] = i1; idx3[(size_t)g * 3 + 2] = i2;
    w3[(size_t)g * 3 + 0] = w0 / ws; w3[(size_t)g * 3 + 1] = w1 / ws; w3[(size_t)g * 3 + 2] = w2 / ws;
}

__global__ void interp3_kernel(const unsigned short* __restrict__ feats2,
                               const int* __restrict__ idx3, const float* __restrict__ w3,
                               unsigned short* __restrict__ dst, int S, int C,
                               int dstStride, int dstOff, int N1, size_t total) {
    size_t i = (size_t)blockIdx.x * 256 + threadIdx.x;
    if (i >= total) return;
    size_t g = i / C; int c = (int)(i % C);
    int b = (int)(g / N1);
    const int* id = &idx3[g * 3];
    const float* w = &w3[g * 3];
    float v = w[0] * bf2f(feats2[((size_t)b * S + id[0]) * C + c])
            + w[1] * bf2f(feats2[((size_t)b * S + id[1]) * C + c])
            + w[2] * bf2f(feats2[((size_t)b * S + id[2]) * C + c]);
    dst[g * (size_t)dstStride + dstOff + c] = f2bf(v);
}

// out[b,c,n] = relu(scale*x+shift), transposed f32 output
__global__ void head_out(const unsigned short* __restrict__ X,
                         const float* __restrict__ scale, const float* __restrict__ shift,
                         float* __restrict__ out, int Npts, int C, size_t total) {
    size_t i = (size_t)blockIdx.x * 256 + threadIdx.x;
    if (i >= total) return;
    size_t row = i / C; int c = (int)(i % C);
    int b = (int)(row / Npts), n = (int)(row % Npts);
    float v = scale[c] * bf2f(X[i]) + shift[c];
    out[((size_t)b * C + c) * Npts + n] = fmaxf(v, 0.f);
}

// ------------------------------ host side ----------------------------------
struct Layer {
    const float *W, *b, *beta, *gamma;
    int cin, cout;
    unsigned short* wbf;   // [pad32(cin), cout] bf16, zero-padded rows
};

static inline int pad32(int x) { return (x + 31) & ~31; }

static inline Layer mkLayer(void* const* d_in, int base, int cin, int cout) {
    Layer L;
    L.W     = (const float*)d_in[base + 0];
    L.b     = (const float*)d_in[base + 1];
    L.beta  = (const float*)d_in[base + 2];
    L.gamma = (const float*)d_in[base + 3];
    L.cin = cin; L.cout = cout; L.wbf = nullptr;
    return L;
}

// Runs a GEMM chain with fused-BN bookkeeping. Returns the final *raw* GEMM
// output; *outScale/*outShift give its pending BN scale/shift (to be applied
// by the consumer).
static const unsigned short* run_mlp(hipStream_t st, const Layer* Ls, const int* li, int nl,
                                     const unsigned short* in, int inKp, int rows,
                                     const float* inScale, const float* inShift,
                                     unsigned short* b0, unsigned short* b1,
                                     float* partS, float* partQ,
                                     float* scaleBase, float* shiftBase,
                                     const float** outScale, const float** outShift) {
    const unsigned short* src = in;
    int kp = inKp;
    const float* sIn = inScale;
    const float* hIn = inShift;
    for (int i = 0; i < nl; ++i) {
        const Layer& L = Ls[li[i]];
        unsigned short* dst = (src == b0) ? b1 : b0;
        unsigned gx = (unsigned)(rows / 64);
        float* scl = scaleBase + (size_t)li[i] * 1024;
        float* sht = shiftBase + (size_t)li[i] * 1024;
        dim3 g(gx, (unsigned)(L.cout / 32));
        gemm_bias_wmma<<<g, dim3(128), 0, st>>>(src, L.wbf, L.b, sIn, hIn,
                                                dst, partS, partQ, rows, kp, L.cout);
        bn_finalize<<<dim3((unsigned)L.cout), dim3(256), 0, st>>>(
            partS, partQ, (int)gx, L.cout, rows, L.gamma, L.beta, scl, sht);
        src = dst;
        kp = L.cout;
        sIn = scl; hIn = sht;
    }
    *outScale = sIn; *outShift = hIn;
    return src;
}

extern "C" void kernel_launch(void* const* d_in, const int* in_sizes, int n_in,
                              void* d_out, int out_size, void* d_ws, size_t ws_size,
                              hipStream_t stream) {
    (void)in_sizes; (void)n_in; (void)out_size; (void)ws_size;

    // ---- workspace carve (deterministic bump allocator) ----
    char* wsp = (char*)d_ws;
    size_t off = 0;
    auto alloc = [&](size_t bytes) -> void* {
        off = (off + 255) & ~(size_t)255;
        void* p = wsp + off;
        off += bytes;
        return p;
    };

    float* xyz0 = (float*)alloc((size_t)BB * NN * 3 * 4);
    float* xyz1 = (float*)alloc((size_t)BB * S1P * 3 * 4);
    float* xyz2 = (float*)alloc((size_t)BB * S2P * 3 * 4);
    int*   fidx1 = (int*)alloc((size_t)BB * S1P * 4);
    int*   fidx2 = (int*)alloc((size_t)BB * S2P * 4);
    int*   bidx  = (int*)alloc((size_t)BB * S1P * 128 * 4);
    unsigned short* l1feat = (unsigned short*)alloc((size_t)BB * S1P * 320 * 2);
    unsigned short* l2feat = (unsigned short*)alloc((size_t)BB * S2P * 640 * 2);
    unsigned short* l3     = (unsigned short*)alloc((size_t)BB * 1024 * 2);
    unsigned short* l2p    = (unsigned short*)alloc((size_t)BB * S2P * 256 * 2);
    unsigned short* l1p    = (unsigned short*)alloc((size_t)BB * S1P * 128 * 2);
    int*   idx3a = (int*)alloc((size_t)BB * S1P * 3 * 4);
    float* w3a   = (float*)alloc((size_t)BB * S1P * 3 * 4);
    int*   idx3b = (int*)alloc((size_t)BB * NN * 3 * 4);
    float* w3b   = (float*)alloc((size_t)BB * NN * 3 * 4);
    // fused-BN bookkeeping: per-block partials + per-layer scale/shift
    float* partS = (float*)alloc((size_t)1048576 * 4);
    float* partQ = (float*)alloc((size_t)1048576 * 4);
    float* scaleArr = (float*)alloc((size_t)31 * 1024 * 4);
    float* shiftArr = (float*)alloc((size_t)31 * 1024 * 4);
    // ping-pong activation buffers (sized for SA1 branch-3 / SA2 group input)
    unsigned short* buf0 = (unsigned short*)alloc((size_t)524288 * 96 * 2);
    unsigned short* buf1 = (unsigned short*)alloc((size_t)524288 * 128 * 2);

    // ---- layer table (JAX pytree order: fp1, fp2, fp3, head, sa1, sa2, sa3) ----
    Layer Ls[31]; int nL = 0;
    int fp1i[3], fp2i[2], fp3i[2], headi, sa1i[3][3], sa2i[3][3], sa3i[3];
    {
        int fp1d[4] = {131, 128, 128, 128};
        for (int i = 0; i < 3; ++i) { Ls[nL] = mkLayer(d_in, 1 + 4 * i, fp1d[i], fp1d[i + 1]); fp1i[i] = nL++; }
        Ls[nL] = mkLayer(d_in, 13, 576, 256); fp2i[0] = nL++;
        Ls[nL] = mkLayer(d_in, 17, 256, 128); fp2i[1] = nL++;
        Ls[nL] = mkLayer(d_in, 21, 1664, 256); fp3i[0] = nL++;
        Ls[nL] = mkLayer(d_in, 25, 256, 256);  fp3i[1] = nL++;
        Ls[nL] = mkLayer(d_in, 29, 128, 128);  headi = nL++;
        int sa1d[3][4] = {{3, 32, 32, 64}, {3, 64, 64, 128}, {3, 64, 96, 128}};
        int base = 33;
        for (int j = 0; j < 3; ++j)
            for (int i = 0; i < 3; ++i) { Ls[nL] = mkLayer(d_in, base, sa1d[j][i], sa1d[j][i + 1]); sa1i[j][i] = nL++; base += 4; }
        int sa2d[3][4] = {{323, 64, 64, 128}, {323, 128, 128, 256}, {323, 128, 128, 256}};
        for (int j = 0; j < 3; ++j)
            for (int i = 0; i < 3; ++i) { Ls[nL] = mkLayer(d_in, base, sa2d[j][i], sa2d[j][i + 1]); sa2i[j][i] = nL++; base += 4; }
        int sa3d[4] = {643, 256, 512, 1024};
        for (int i = 0; i < 3; ++i) { Ls[nL] = mkLayer(d_in, base + 4 * i, sa3d[i], sa3d[i + 1]); sa3i[i] = nL++; }
    }
    // convert all weights to bf16, zero-padded to pad32(cin) rows
    for (int i = 0; i < nL; ++i) {
        size_t n = (size_t)pad32(Ls[i].cin) * Ls[i].cout;
        Ls[i].wbf = (unsigned short*)alloc(n * 2);
        conv_weight<<<dim3((unsigned)((n + 255) / 256)), dim3(256), 0, stream>>>(
            Ls[i].W, Ls[i].wbf, Ls[i].cin, Ls[i].cout, n);
    }

    const float* x_in = (const float*)d_in[0];
    const float *oS, *oH;   // pending scale/shift of an MLP's last layer

    // ---- l0: transpose input to [B,N,3] ----
    {
        size_t tot = (size_t)BB * NN * 3;
        transpose_in<<<dim3((unsigned)((tot + 255) / 256)), dim3(256), 0, stream>>>(x_in, xyz0, NN, tot);
    }

    // ---- SA1 ----
    fps_kernel<<<dim3(BB), dim3(256), 0, stream>>>(xyz0, NN, S1P, fidx1);
    gather_xyz<<<dim3((BB * S1P + 255) / 256), dim3(256), 0, stream>>>(
        xyz0, fidx1, xyz1, NN, S1P, BB * S1P);
    {
        const float rad[3] = {0.1f, 0.2f, 0.4f};
        const int   ks[3]  = {16, 32, 128};
        const int   cout[3] = {64, 128, 128};
        const int   offs[3] = {0, 64, 192};
        for (int j = 0; j < 3; ++j) {
            int K = ks[j]; float r2 = rad[j] * rad[j];
            int totq = BB * S1P;
            ball_query<<<dim3((totq + 255) / 256), dim3(256), 0, stream>>>(
                xyz0, xyz1, NN, S1P, r2, K, bidx, totq);
            int rows = totq * K;
            int cpad = pad32(3);   // 32
            size_t tot = (size_t)rows * cpad;
            group_kernel<<<dim3((unsigned)((tot + 255) / 256)), dim3(256), 0, stream>>>(
                xyz0, xyz1, (const unsigned short*)nullptr, bidx, buf0, NN, S1P, K, 0, cpad, tot);
            const unsigned short* o = run_mlp(stream, Ls, sa1i[j], 3, buf0, cpad, rows,
                                              nullptr, nullptr, buf0, buf1,
                                              partS, partQ, scaleArr, shiftArr, &oS, &oH);
            size_t pt = (size_t)totq * cout[j];
            maxpool_k<<<dim3((unsigned)((pt + 255) / 256)), dim3(256), 0, stream>>>(
                o, oS, oH, l1feat, K, cout[j], 320, offs[j], pt);
        }
    }

    // ---- SA2 ----
    fps_kernel<<<dim3(BB), dim3(256), 0, stream>>>(xyz1, S1P, S2P, fidx2);
    gather_xyz<<<dim3((BB * S2P + 255) / 256), dim3(256), 0, stream>>>(
        xyz1, fidx2, xyz2, S1P, S2P, BB * S2P);
    {
        const float rad[3] = {0.2f, 0.4f, 0.8f};
        const int   ks[3]  = {32, 64, 128};
        const int   cout[3] = {128, 256, 256};
        const int   offs[3] = {0, 128, 384};
        for (int j = 0; j < 3; ++j) {
            int K = ks[j]; float r2 = rad[j] * rad[j];
            int totq = BB * S2P;
            ball_query<<<dim3((totq + 255) / 256), dim3(256), 0, stream>>>(
                xyz1, xyz2, S1P, S2P, r2, K, bidx, totq);
            int rows = totq * K;
            int cpad = pad32(323); // 352
            size_t tot = (size_t)rows * cpad;
            group_kernel<<<dim3((unsigned)((tot + 255) / 256)), dim3(256), 0, stream>>>(
                xyz1, xyz2, l1feat, bidx, buf0, S1P, S2P, K, 320, cpad, tot);
            const unsigned short* o = run_mlp(stream, Ls, sa2i[j], 3, buf0, cpad, rows,
                                              nullptr, nullptr, buf0, buf1,
                                              partS, partQ, scaleArr, shiftArr, &oS, &oH);
            size_t pt = (size_t)totq * cout[j];
            maxpool_k<<<dim3((unsigned)((pt + 255) / 256)), dim3(256), 0, stream>>>(
                o, oS, oH, l2feat, K, cout[j], 640, offs[j], pt);
        }
    }

    // ---- SA3 (group-all) ----
    {
        int rows = BB * S2P;                   // 1024
        int cpad = pad32(643);                 // 672
        size_t tot = (size_t)rows * cpad;
        concat_feats_xyz<<<dim3((unsigned)((tot + 255) / 256)), dim3(256), 0, stream>>>(
            l2feat, 640, xyz2, buf0, cpad, tot);
        const unsigned short* o = run_mlp(stream, Ls, sa3i, 3, buf0, cpad, rows,
                                          nullptr, nullptr, buf0, buf1,
                                          partS, partQ, scaleArr, shiftArr, &oS, &oH);
        size_t pt = (size_t)BB * 1024;
        maxpool_k<<<dim3((unsigned)((pt + 255) / 256)), dim3(256), 0, stream>>>(
            o, oS, oH, l3, S2P, 1024, 1024, 0, pt);
    }

    // ---- FP3: broadcast l3 + concat l2 -> MLP (Cin=1664, already %32) ----
    {
        int rows = BB * S2P;
        size_t cp = (size_t)rows * 640;
        copy_bf16<<<dim3((unsigned)((cp + 255) / 256)), dim3(256), 0, stream>>>(
            l2feat, 640, buf0, 1664, 0, 640, cp);
        size_t bc = (size_t)rows * 1024;
        bcast_rows<<<dim3((unsigned)((bc + 255) / 256)), dim3(256), 0, stream>>>(
            l3, buf0, 1664, 640, 1024, S2P, bc);
        const unsigned short* o = run_mlp(stream, Ls, fp3i, 2, buf0, 1664, rows,
                                          nullptr, nullptr, buf0, buf1,
                                          partS, partQ, scaleArr, shiftArr, &oS, &oH);
        size_t cc = (size_t)rows * 256;
        apply_copy<<<dim3((unsigned)((cc + 255) / 256)), dim3(256), 0, stream>>>(
            o, oS, oH, l2p, 256, cc);
    }

    // ---- FP2: 3-NN interp l2p -> l1 level (Cin=576, already %32) ----
    {
        int totq = BB * S1P;  // 4096
        three_nn_kernel<<<dim3((totq + 255) / 256), dim3(256), 0, stream>>>(
            xyz1, xyz2, S1P, S2P, idx3a, w3a, totq);
        size_t cp = (size_t)totq * 320;
        copy_bf16<<<dim3((unsigned)((cp + 255) / 256)), dim3(256), 0, stream>>>(
            l1feat, 320, buf0, 576, 0, 320, cp);
        size_t it = (size_t)totq * 256;
        interp3_kernel<<<dim3((unsigned)((it + 255) / 256)), dim3(256), 0, stream>>>(
            l2p, idx3a, w3a, buf0, S2P, 256, 576, 320, S1P, it);
        const unsigned short* o = run_mlp(stream, Ls, fp2i, 2, buf0, 576, totq,
                                          nullptr, nullptr, buf0, buf1,
                                          partS, partQ, scaleArr, shiftArr, &oS, &oH);
        size_t cc = (size_t)totq * 128;
        apply_copy<<<dim3((unsigned)((cc + 255) / 256)), dim3(256), 0, stream>>>(
            o, oS, oH, l1p, 128, cc);
    }

    // ---- FP1: 3-NN interp l1p -> l0 level (Cin=131 -> pad 160) ----
    const unsigned short* ofp1;
    {
        int totq = BB * NN;   // 32768
        int cpad = pad32(131); // 160
        three_nn_kernel<<<dim3((totq + 255) / 256), dim3(256), 0, stream>>>(
            xyz0, xyz1, NN, S1P, idx3b, w3b, totq);
        size_t xb = (size_t)totq * 3;
        xyz_bf16<<<dim3((unsigned)((xb + 255) / 256)), dim3(256), 0, stream>>>(
            xyz0, buf0, cpad, 0, xb);
        size_t it = (size_t)totq * 128;
        interp3_kernel<<<dim3((unsigned)((it + 255) / 256)), dim3(256), 0, stream>>>(
            l1p, idx3b, w3b, buf0, S1P, 128, cpad, 3, NN, it);
        size_t zc = (size_t)totq * (cpad - 131);
        zero_cols<<<dim3((unsigned)((zc + 255) / 256)), dim3(256), 0, stream>>>(
            buf0, cpad, 131, cpad - 131, zc);
        ofp1 = run_mlp(stream, Ls, fp1i, 3, buf0, cpad, totq,
                       nullptr, nullptr, buf0, buf1,
                       partS, partQ, scaleArr, shiftArr, &oS, &oH);
    }

    // ---- head: GEMM (fused BN on input) + BN + ReLU -> [B, OUT, N] f32 ----
    {
        int rows = BB * NN;
        unsigned short* hdst = (ofp1 == buf0) ? buf1 : buf0;
        const Layer& H = Ls[headi];
        unsigned gx = (unsigned)(rows / 64);
        float* scl = scaleArr + (size_t)headi * 1024;
        float* sht = shiftArr + (size_t)headi * 1024;
        dim3 g(gx, (unsigned)(128 / 32));
        gemm_bias_wmma<<<g, dim3(128), 0, stream>>>(ofp1, H.wbf, H.b, oS, oH,
                                                    hdst, partS, partQ, rows, 128, 128);
        bn_finalize<<<dim3(128), dim3(256), 0, stream>>>(
            partS, partQ, (int)gx, 128, rows, H.gamma, H.beta, scl, sht);
        size_t ht = (size_t)rows * 128;
        head_out<<<dim3((unsigned)((ht + 255) / 256)), dim3(256), 0, stream>>>(
            hdst, scl, sht, (float*)d_out, NN, 128, ht);
    }
}
